// RecurrentModel_28939489641200
// MI455X (gfx1250) — compile-verified, matrix operand order checked by reading, not run on hardware
//
#include <hip/hip_runtime.h>
#include <cstdint>
#include <cstddef>

#define S_LEN 32768
#define E_DIM 512
#define H_DIM 512
#define T_CLS 64
#define CH    8        // scan xW chunk rows per TDM transfer

typedef __attribute__((ext_vector_type(16))) __bf16 v16bf;
typedef __attribute__((ext_vector_type(8)))  float  v8f;
typedef unsigned int u32x4 __attribute__((ext_vector_type(4)));
typedef int          i32x4 __attribute__((ext_vector_type(4)));
typedef int          i32x8 __attribute__((ext_vector_type(8)));

#if defined(__has_builtin)
#if __has_builtin(__builtin_amdgcn_tensor_load_to_lds) && __has_builtin(__builtin_amdgcn_s_wait_tensorcnt)
#define HAVE_TDM 1
#endif
#endif
#ifndef HAVE_TDM
#define HAVE_TDM 0
#endif

// ---------------------------------------------------------------------------
// bf16 pack (RNE); unpack at use-site is pure bit-ops (shl / and).
// ---------------------------------------------------------------------------
__device__ __forceinline__ unsigned pack_bf16x2(float a, float b) {
  unsigned ua = __float_as_uint(a), ub = __float_as_uint(b);
  ua += 0x7fffu + ((ua >> 16) & 1u);
  ub += 0x7fffu + ((ub >> 16) & 1u);
  return (ua >> 16) | (ub & 0xffff0000u);
}

// ---------------------------------------------------------------------------
// TDM descriptor build + issue: load `nrows` rows of a [S_LEN x 512] fp32
// tensor into LDS. D# per ISA 8.3/8.4: group0 = {count, lds_addr, global_addr,
// type=2}; group1 = data_size/dims/tile/strides. 2D tile -> groups 2/3 zero.
// ---------------------------------------------------------------------------
__device__ __forceinline__ void tdm_load_rows_512f(const float* gsrc,
                                                   unsigned lds_off, int nrows) {
#if HAVE_TDM
  const unsigned long long ga = (unsigned long long)(size_t)gsrc;
  u32x4 g0 = {1u,                                   // count=1, user descriptor
              lds_off,                              // LDS byte address
              (unsigned)(ga & 0xffffffffu),         // global_addr[31:0]
              (unsigned)((ga >> 32) & 0x01ffffffu) | (2u << 30)};  // hi | type=2
  i32x8 g1 = {(int)(2u << 16),                      // data_size = 4B
              (int)((unsigned)(E_DIM & 0xffff) << 16),            // tensor_dim0 lo
              (int)((((unsigned)E_DIM) >> 16) |
                    (((unsigned)S_LEN & 0xffffu) << 16)),          // td0 hi | td1 lo
              (int)(((((unsigned)S_LEN) >> 16) & 0xffffu) |
                    ((unsigned)E_DIM << 16)),                      // td1 hi | tile_dim0
              (int)(nrows & 0xffff),                // tile_dim1 (tile_dim2=0)
              (int)E_DIM,                           // tensor_dim0_stride lo32
              0, 0};
  i32x4 gz = {0, 0, 0, 0};
#if defined(__clang_major__) && (__clang_major__ >= 23)
  i32x8 gz8 = {0, 0, 0, 0, 0, 0, 0, 0};
  __builtin_amdgcn_tensor_load_to_lds(g0, g1, gz, gz, gz8, 0);
#else
  __builtin_amdgcn_tensor_load_to_lds(g0, g1, gz, gz, 0);
#endif
#else
  (void)gsrc; (void)lds_off; (void)nrows;
#endif
}

// ---------------------------------------------------------------------------
// Kernel 2 (placed first so the disasm snippet shows its hot loop):
// sequential bidirectional scan. One workgroup (32 wave32) per direction.
// 2 threads per output feature (wave32 pair, combined via shfl_xor).
// Hybrid W_hh residency: 128 k-values (64 packed-bf16 dwords) in VGPRs,
// 128 k-values in LDS (512 x 128 dwords = 256 KB of the 320 KB WGP LDS) --
// keeps per-thread VGPRs ~90 so no spills even at CU-mode wave occupancy.
// h lives in LDS (float4 broadcast reads). xW chunks are streamed into
// double-buffered LDS by the Tensor Data Mover, overlapped with compute.
// h' = tanh(xw + h.W_hh[f,:] + b_hh[f]); hidden states stored bf16.
// ---------------------------------------------------------------------------
__global__ __launch_bounds__(1024, 1) void k_rnn_scan(
    const float* __restrict__ xwf, const float* __restrict__ xwb,
    const float* __restrict__ Whhf, const float* __restrict__ bhhf,
    const float* __restrict__ Whhb, const float* __restrict__ bhhb,
    __bf16* __restrict__ hf, __bf16* __restrict__ hb) {
  const int d = blockIdx.x;
  const float* xw  = d ? xwb  : xwf;
  const float* Whh = d ? Whhb : Whhf;
  const float* bhh = d ? bhhb : bhhf;
  __bf16*      hs  = d ? hb   : hf;

  const int tid   = threadIdx.x;
  const int f     = tid >> 1;       // output feature
  const int khalf = tid & 1;        // which 256-wide K slice
  const int kbase = khalf * 256;

  __shared__ float    sh_h[H_DIM];                //   2 KB
  __shared__ float    sxw[2][CH * H_DIM];         //  32 KB (TDM double buffer)
  __shared__ unsigned swt[H_DIM * 128];           // 256 KB packed-bf16 weights

  // Stage weights: k in [kbase, kbase+128) -> 64 dwords in VGPRs,
  //                k in [kbase+128, kbase+256) -> 64 dwords in LDS.
  unsigned wreg[64];
  {
    const float* wrA = Whh + (size_t)f * H_DIM + kbase;
#pragma unroll
    for (int i = 0; i < 64; ++i) wreg[i] = pack_bf16x2(wrA[2 * i], wrA[2 * i + 1]);
    const float* wrB = wrA + 128;
    unsigned* sl = &swt[f * 128 + khalf * 64];
    for (int i = 0; i < 64; ++i) sl[i] = pack_bf16x2(wrB[2 * i], wrB[2 * i + 1]);
  }
  if (tid < H_DIM) sh_h[tid] = 0.0f;
  const float bias = bhh[f];

#if HAVE_TDM
  if (tid == 0) {  // prime chunk 0 (EXEC ignored by TDM; one op per WG)
    tdm_load_rows_512f(xw, (unsigned)(size_t)&sxw[0][0], CH);
    __builtin_amdgcn_s_wait_tensorcnt(0);
  }
#endif
  __syncthreads();

  const uint4*  wl  = (const uint4*)&swt[f * 128 + khalf * 64];
  const float4* h4a = (const float4*)&sh_h[kbase];
  const float4* h4b = (const float4*)&sh_h[kbase + 128];

#pragma unroll 1
  for (int t0 = 0; t0 < S_LEN; t0 += CH) {
    const int buf = (t0 / CH) & 1;
#if HAVE_TDM
    if (tid == 0 && (t0 + CH) < S_LEN)   // async: fetch next chunk
      tdm_load_rows_512f(xw + (size_t)(t0 + CH) * H_DIM,
                         (unsigned)(size_t)&sxw[buf ^ 1][0], CH);
#else
    for (int i = tid; i < CH * H_DIM; i += 1024)
      sxw[buf][i] = xw[(size_t)t0 * H_DIM + i];
    __syncthreads();
#endif

#pragma unroll 1
    for (int tt = 0; tt < CH; ++tt) {
      float acc = 0.0f;
#pragma unroll
      for (int i = 0; i < 32; ++i) {            // register-resident half
        const float4  hh = h4a[i];
        const unsigned w0 = wreg[2 * i], w1 = wreg[2 * i + 1];
        acc = fmaf(__uint_as_float(w0 << 16),          hh.x, acc);
        acc = fmaf(__uint_as_float(w0 & 0xffff0000u),  hh.y, acc);
        acc = fmaf(__uint_as_float(w1 << 16),          hh.z, acc);
        acc = fmaf(__uint_as_float(w1 & 0xffff0000u),  hh.w, acc);
      }
#pragma unroll
      for (int i = 0; i < 16; ++i) {            // LDS-resident half (b128 loads)
        const uint4  w4 = wl[i];
        const float4 ha = h4b[2 * i], hbv = h4b[2 * i + 1];
        acc = fmaf(__uint_as_float(w4.x << 16),          ha.x, acc);
        acc = fmaf(__uint_as_float(w4.x & 0xffff0000u),  ha.y, acc);
        acc = fmaf(__uint_as_float(w4.y << 16),          ha.z, acc);
        acc = fmaf(__uint_as_float(w4.y & 0xffff0000u),  ha.w, acc);
        acc = fmaf(__uint_as_float(w4.z << 16),          hbv.x, acc);
        acc = fmaf(__uint_as_float(w4.z & 0xffff0000u),  hbv.y, acc);
        acc = fmaf(__uint_as_float(w4.w << 16),          hbv.z, acc);
        acc = fmaf(__uint_as_float(w4.w & 0xffff0000u),  hbv.w, acc);
      }
      acc += __shfl_xor(acc, 1);          // combine the two K halves (wave32)
      __syncthreads();                    // all h_{t-1} reads complete
      if (khalf == 0) {
        const float hv = tanhf(acc + sxw[buf][tt * H_DIM + f] + bias);
        sh_h[f] = hv;
        hs[(size_t)(t0 + tt) * H_DIM + f] = (__bf16)hv;
      }
      __syncthreads();                    // h_t visible
    }
#if HAVE_TDM
    if (tid == 0) __builtin_amdgcn_s_wait_tensorcnt(0);
    __syncthreads();
#endif
  }
}

// ---------------------------------------------------------------------------
// WMMA 16-bit A/B fragment load from LDS, per ISA 7.12.2 layout:
//   lanes 0-15 : row/col = lane,     VGPR v holds K = {2v | 8+2v} (+0)
//   lanes 16-31: row/col = lane-16,  same K map shifted by +8
// ---------------------------------------------------------------------------
__device__ __forceinline__ void load_frag_bf16(const __bf16* base, int ld,
                                               int lane, int kOff, v16bf& f) {
  const int r  = lane & 15;
  const int kb = ((lane & 16) ? 8 : 0) + kOff;
#pragma unroll
  for (int v = 0; v < 8; ++v) {
    const int k0 = ((v < 4) ? (2 * v) : (8 + 2 * v)) + kb;
    f[2 * v]     = base[r * ld + k0];
    f[2 * v + 1] = base[r * ld + k0 + 1];
  }
}

// ---------------------------------------------------------------------------
// Kernel 1: embedding gather + input projection for both directions.
// out[d][t, n] = emb[row_d(t)] . W_ih_d[n, :] + b_ih_d[n]
// Block tile: M=64 (seq) x N=32 (hidden), 8 waves -> 4x2 grid of 16x16 WMMA
// tiles, K staged 64 at a time in LDS (2 wmma per wave per stage).
// ---------------------------------------------------------------------------
__global__ __launch_bounds__(256) void k_embed_project(
    const int* __restrict__ tokens, const float* __restrict__ emb,
    const float* __restrict__ Wf, const float* __restrict__ bf_,
    const float* __restrict__ Wb, const float* __restrict__ bb_,
    float* __restrict__ xwf, float* __restrict__ xwb) {
  const int d = blockIdx.z;
  const float* W    = d ? Wb  : Wf;
  const float* bias = d ? bb_ : bf_;
  float*       out  = d ? xwb : xwf;

  __shared__ __bf16 As[64][72];   // +8 pad: bank-conflict-free
  __shared__ __bf16 Bs[32][72];

  const int tid  = threadIdx.x;
  const int lane = tid & 31;
  const int wave = tid >> 5;
  const int mw = wave >> 1, nw = wave & 1;
  const int m0 = blockIdx.x * 64;
  const int n0 = blockIdx.y * 32;

  // A loader: 4 threads/row, 16 contiguous floats each; token cached once.
  const int arow  = tid >> 2;
  const int akoff = (tid & 3) * 16;
  const int t_glob = m0 + arow;
  const int src    = d ? (S_LEN - 1 - t_glob) : t_glob;
  const float* erow = emb + (size_t)tokens[src] * E_DIM;

  // B loader: W[n0+brow, :] row, 8 contiguous floats each.
  const int brow  = tid >> 3;
  const int bkoff = (tid & 7) * 8;
  const float* wrow = W + (size_t)(n0 + brow) * E_DIM;

  v8f acc = {0.f, 0.f, 0.f, 0.f, 0.f, 0.f, 0.f, 0.f};

  for (int kb = 0; kb < E_DIM; kb += 64) {
#pragma unroll
    for (int i = 0; i < 16; ++i)
      As[arow][akoff + i] = (__bf16)erow[kb + akoff + i];
#pragma unroll
    for (int i = 0; i < 8; ++i)
      Bs[brow][bkoff + i] = (__bf16)wrow[kb + bkoff + i];
    if (kb + 64 < E_DIM) __builtin_prefetch(erow + kb + 64 + akoff, 0, 3);
    __syncthreads();

    v16bf a0, b0, a1, b1;
    load_frag_bf16(&As[mw * 16][0], 72, lane, 0, a0);
    load_frag_bf16(&Bs[nw * 16][0], 72, lane, 0, b0);
    acc = __builtin_amdgcn_wmma_f32_16x16x32_bf16(false, a0, false, b0,
                                                  (short)0, acc, false, false);
    load_frag_bf16(&As[mw * 16][0], 72, lane, 32, a1);
    load_frag_bf16(&Bs[nw * 16][0], 72, lane, 32, b1);
    acc = __builtin_amdgcn_wmma_f32_16x16x32_bf16(false, a1, false, b1,
                                                  (short)0, acc, false, false);
    __syncthreads();
  }

  // C/D layout: VGPR r -> lanes 0-15: (M=r, N=lane); lanes 16-31: (M=r+8).
#pragma unroll
  for (int r = 0; r < 8; ++r) {
    const int row = mw * 16 + ((lane & 16) ? (r + 8) : r);
    const int col = nw * 16 + (lane & 15);
    out[(size_t)(m0 + row) * H_DIM + (n0 + col)] = acc[r] + bias[n0 + col];
  }
}

// ---------------------------------------------------------------------------
// Kernel 3: classifier GEMM (WMMA) + bias + fused log-softmax.
// Block: M=32 seq rows x N=64 (=T) classes; 8 waves -> 2x4 tiles; K=1024.
// hidden[s] = [hf[s], hb[S-1-s]] (hb stored in scan order).
// ---------------------------------------------------------------------------
__global__ __launch_bounds__(256) void k_classify(
    const __bf16* __restrict__ hf, const __bf16* __restrict__ hb,
    const float* __restrict__ Wc, const float* __restrict__ bc,
    float* __restrict__ out) {
  __shared__ __bf16 As[32][40];
  __shared__ __bf16 Bs[64][40];
  __shared__ float  Ls[32][66];

  const int tid = threadIdx.x, lane = tid & 31, wave = tid >> 5;
  const int mw = wave >> 2, nw = wave & 3;
  const int m0 = blockIdx.x * 32;

  const int arow = tid >> 3, akoff = (tid & 7) * 4;
  const int brow = tid >> 2, bkoff = (tid & 3) * 8;
  const int s_a = m0 + arow;

  v8f acc = {0.f, 0.f, 0.f, 0.f, 0.f, 0.f, 0.f, 0.f};

  for (int kb = 0; kb < 2 * H_DIM; kb += 32) {
#pragma unroll
    for (int i = 0; i < 4; ++i) {
      const int j = kb + akoff + i;
      As[arow][akoff + i] =
          (j < H_DIM) ? hf[(size_t)s_a * H_DIM + j]
                      : hb[(size_t)(S_LEN - 1 - s_a) * H_DIM + (j - H_DIM)];
    }
#pragma unroll
    for (int i = 0; i < 8; ++i)
      Bs[brow][bkoff + i] = (__bf16)Wc[(size_t)brow * (2 * H_DIM) + kb + bkoff + i];
    __syncthreads();

    v16bf a, b;
    load_frag_bf16(&As[mw * 16][0], 40, lane, 0, a);
    load_frag_bf16(&Bs[nw * 16][0], 40, lane, 0, b);
    acc = __builtin_amdgcn_wmma_f32_16x16x32_bf16(false, a, false, b,
                                                  (short)0, acc, false, false);
    __syncthreads();
  }

#pragma unroll
  for (int r = 0; r < 8; ++r) {
    const int row = mw * 16 + ((lane & 16) ? (r + 8) : r);
    const int col = nw * 16 + (lane & 15);
    Ls[row][col] = acc[r] + bc[col];
  }
  __syncthreads();

  if (tid < 32) {                         // one row of 64 logits per thread
    float mx = -1e30f;
    for (int c = 0; c < T_CLS; ++c) mx = fmaxf(mx, Ls[tid][c]);
    float se = 0.0f;
    for (int c = 0; c < T_CLS; ++c) se += expf(Ls[tid][c] - mx);
    const float lse = mx + logf(se);
    float* orow = out + (size_t)(m0 + tid) * T_CLS;
    for (int c = 0; c < T_CLS; ++c) orow[c] = Ls[tid][c] - lse;
  }
}

// ---------------------------------------------------------------------------
extern "C" void kernel_launch(void* const* d_in, const int* in_sizes, int n_in,
                              void* d_out, int out_size, void* d_ws, size_t ws_size,
                              hipStream_t stream) {
  (void)in_sizes; (void)n_in; (void)out_size; (void)ws_size;
  const int*   tokens = (const int*)d_in[0];
  const float* emb    = (const float*)d_in[1];
  const float* Wihf   = (const float*)d_in[2];
  const float* Whhf   = (const float*)d_in[3];
  const float* bihf   = (const float*)d_in[4];
  const float* bhhf   = (const float*)d_in[5];
  const float* Wihb   = (const float*)d_in[6];
  const float* Whhb   = (const float*)d_in[7];
  const float* bihb   = (const float*)d_in[8];
  const float* bhhb   = (const float*)d_in[9];
  const float* Wcls   = (const float*)d_in[10];
  const float* bcls   = (const float*)d_in[11];
  float* out = (float*)d_out;

  char* ws = (char*)d_ws;
  const size_t f32Plane  = (size_t)S_LEN * H_DIM * sizeof(float);   // 64 MB
  const size_t bf16Plane = (size_t)S_LEN * H_DIM * sizeof(__bf16);  // 32 MB
  float*  xwf = (float*)ws;
  float*  xwb = (float*)(ws + f32Plane);
  __bf16* hfw = (__bf16*)(ws + 2 * f32Plane);
  __bf16* hbw = (__bf16*)(ws + 2 * f32Plane + bf16Plane);

  dim3 g1(S_LEN / 64, H_DIM / 32, 2), b1(256);
  k_embed_project<<<g1, b1, 0, stream>>>(tokens, emb, Wihf, bihf, Wihb, bihb,
                                         xwf, xwb);
  k_rnn_scan<<<dim3(2), dim3(1024), 0, stream>>>(xwf, xwb, Whhf, bhhf,
                                                 Whhb, bhhb, hfw, hbw);
  k_classify<<<dim3(S_LEN / 32), dim3(256), 0, stream>>>(hfw, hbw, Wcls, bcls,
                                                         out);
}
